// BatchedBitNetFFN_41815801594114
// MI455X (gfx1250) — 8-bit matrix operand order repaired and checked by reading, compiled, not run on hardware
//
#include <hip/hip_runtime.h>
#include <hip/hip_bf16.h>


typedef int v8i __attribute__((ext_vector_type(8)));
typedef unsigned int u32x4 __attribute__((ext_vector_type(4)));
typedef int i32x8 __attribute__((ext_vector_type(8)));
typedef int i32x4 __attribute__((ext_vector_type(4)));

#if defined(__has_builtin)
#if __has_builtin(__builtin_amdgcn_tensor_load_to_lds) && \
    __has_builtin(__builtin_amdgcn_s_wait_tensorcnt)
#define USE_TDM 1
#endif
#endif
#ifndef USE_TDM
#define USE_TDM 0
#endif

static constexpr int kE = 16;
static constexpr int kC = 4096;
static constexpr int kD = 768;
static constexpr int kF = 2048;

__global__ void __launch_bounds__(256) wscale_kernel(const float* __restrict__ w,
                                                     float* __restrict__ scale,
                                                     long long n_per_e) {
  __shared__ float red[256];
  const int e = blockIdx.x;
  const float* we = w + (long long)e * n_per_e;
  float s = 0.f;
  for (long long i = threadIdx.x; i < n_per_e; i += 256) s += fabsf(we[i]);
  red[threadIdx.x] = s;
  __syncthreads();
  for (int off = 128; off > 0; off >>= 1) {
    if ((int)threadIdx.x < off) red[threadIdx.x] += red[threadIdx.x + off];
    __syncthreads();
  }
  if (threadIdx.x == 0) scale[e] = fmaxf(red[0] / (float)n_per_e, 1e-5f);
}

__global__ void __launch_bounds__(256) wquant_t_kernel(
    const float* __restrict__ w, const float* __restrict__ scale,
    signed char* __restrict__ wq, int R, int C) {
  __shared__ signed char T[64][65];
  const int e = blockIdx.z;
  const int c0 = blockIdx.x * 64;
  const int r0 = blockIdx.y * 64;
  const int tid = threadIdx.x;
  const float inv = 1.f / scale[e];
  {
    const int rl = tid >> 2;
    const int cl = (tid & 3) * 16;
    const float* src = w + ((long long)e * R + r0 + rl) * C + c0 + cl;
#pragma unroll
    for (int j = 0; j < 16; j++)
      T[rl][cl + j] =
          (signed char)rintf(fminf(fmaxf(src[j] * inv, -1.f), 1.f));
  }
  __syncthreads();
  {
    const int cl = tid >> 2;
    const int rl = (tid & 3) * 16;
    alignas(16) signed char tmp[16];
#pragma unroll
    for (int j = 0; j < 16; j++) tmp[j] = T[rl + j][cl];
    signed char* dst = wq + ((long long)e * C + c0 + cl) * R + r0 + rl;
    *(int4*)dst = *(const int4*)tmp;
  }
}

__global__ void __launch_bounds__(256) aquant_kernel(const float* __restrict__ x,
                                                     signed char* __restrict__ xq,
                                                     float* __restrict__ xs,
                                                     int L) {
  __shared__ float red[256];
  const long long row = blockIdx.x;
  const float* xr = x + row * (long long)L;
  float m = 0.f;
  for (int i = threadIdx.x; i < L; i += 256) m = fmaxf(m, fabsf(xr[i]));
  red[threadIdx.x] = m;
  __syncthreads();
  for (int off = 128; off > 0; off >>= 1) {
    if ((int)threadIdx.x < off)
      red[threadIdx.x] = fmaxf(red[threadIdx.x], red[threadIdx.x + off]);
    __syncthreads();
  }
  const float s = 127.f / fmaxf(red[0], 1e-5f);
  if (threadIdx.x == 0) xs[row] = s;
  signed char* qr = xq + row * (long long)L;
  for (int i = threadIdx.x; i < L; i += 256)
    qr[i] = (signed char)rintf(fminf(fmaxf(xr[i] * s, -128.f), 127.f));
}

#if USE_TDM
__device__ __forceinline__ unsigned lds_off(const void* p) {
  return (unsigned)(unsigned long long)p;
}

__device__ __forceinline__ void tdm_load_2d(const void* gsrc, unsigned ldsaddr,
                                            unsigned rows, unsigned pitch) {
  const unsigned long long ga = (unsigned long long)gsrc;
  u32x4 g0;
  g0[0] = 1u;
  g0[1] = ldsaddr;
  g0[2] = (unsigned)ga;
  g0[3] = (unsigned)(ga >> 32) | (2u << 30);
  i32x8 g1;
  g1[0] = 0;
  g1[1] = (int)(64u << 16);
  g1[2] = (int)(rows << 16);
  g1[3] = (int)(64u << 16);
  g1[4] = (int)rows;
  g1[5] = (int)pitch;
  g1[6] = 0;
  g1[7] = 0;
  i32x4 z4 = {};
  i32x8 z8 = {};
  __builtin_amdgcn_tensor_load_to_lds(g0, g1, z4, z4, z8, 0);
}
#endif

__device__ __forceinline__ void load_a_tile128(const signed char* __restrict__ gA,
                                               int ldA, signed char* Alds,
                                               int tid) {
#pragma unroll
  for (int i = 0; i < 2; i++) {
    int off = tid * 32 + i * 16;
    int r = off >> 6, c = off & 63;
    *(int4*)(Alds + off) = *(const int4*)(gA + (long long)r * ldA + c);
  }
}

__device__ __forceinline__ void load_b_tile64(const signed char* __restrict__ gB,
                                              int ldB, signed char* Blds,
                                              int tid) {
  int off = tid * 16;
  int r = off >> 6, c = off & 63;
  *(int4*)(Blds + off) = *(const int4*)(gB + (long long)r * ldB + c);
}

__device__ __forceinline__ v8i frag_a(const signed char* Alds, int lane, int m0) {
  v8i a;
  const int m = m0 + (lane & 15);
  const int khalf = (lane >> 4) * 8;
#pragma unroll
  for (int v = 0; v < 8; v++) {
    int kb = (v >> 1) * 16 + (v & 1) * 4 + khalf;
    a[v] = *(const int*)(Alds + m * 64 + kb);
  }
  return a;
}

__device__ __forceinline__ v8i frag_b(const signed char* Blds, int lane, int n0) {
  v8i b;
#pragma unroll
  for (int v = 0; v < 8; v++) {
    int kb = (v >> 1) * 16 + (v & 1) * 4 + (lane >> 4) * 8;
    const signed char* p = Blds + n0 + (lane & 15);
    unsigned w = 0;
#pragma unroll
    for (int j = 0; j < 4; j++)
      w |= (unsigned)(unsigned char)p[(kb + j) * 64] << (8 * j);
    b[v] = (int)w;
  }
  return b;
}

__global__ void __launch_bounds__(256) ffn_gateup_kernel(
    const signed char* __restrict__ xq, const float* __restrict__ xs,
    const signed char* __restrict__ wgq, const signed char* __restrict__ wuq,
    const float* __restrict__ wgs, const float* __restrict__ wus,
    float* __restrict__ h) {
#if USE_TDM
  __shared__ signed char Alds[2][128 * 64];
  __shared__ signed char Bgl[2][64 * 64];
  __shared__ signed char Bul[2][64 * 64];
#else
  __shared__ signed char Alds[1][128 * 64];
  __shared__ signed char Bgl[1][64 * 64];
  __shared__ signed char Bul[1][64 * 64];
#endif

  const int e = blockIdx.z;
  const int m0 = blockIdx.x * 128;
  const int n0 = blockIdx.y * 64;
  const int tid = threadIdx.x;
  const int lane = tid & 31;
  const int wave = tid >> 5;

  const signed char* gA = xq + ((long long)e * kC + m0) * kD;
  const signed char* gG = wgq + (long long)e * kD * kF + n0;
  const signed char* gU = wuq + (long long)e * kD * kF + n0;

  v8i accg[4], accu[4];
#pragma unroll
  for (int t = 0; t < 4; t++) {
    v8i z = {};
    accg[t] = z;
    accu[t] = z;
  }

#if USE_TDM
  if (wave == 0) tdm_load_2d(gA, lds_off(&Alds[0][0]), 128, kD);
  else if (wave == 1) tdm_load_2d(gG, lds_off(&Bgl[0][0]), 64, kF);
  else if (wave == 2) tdm_load_2d(gU, lds_off(&Bul[0][0]), 64, kF);
  if (wave < 3) __builtin_amdgcn_s_wait_tensorcnt(0);
  __syncthreads();
  int cur = 0;
#endif

  for (int kk = 0; kk < kD; kk += 64) {
#if USE_TDM
    const int nxt = cur ^ 1;
    const bool more = (kk + 64) < kD;
    if (more) {
      if (wave == 0)
        tdm_load_2d(gA + kk + 64, lds_off(&Alds[nxt][0]), 128, kD);
      else if (wave == 1)
        tdm_load_2d(gG + (long long)(kk + 64) * kF, lds_off(&Bgl[nxt][0]), 64, kF);
      else if (wave == 2)
        tdm_load_2d(gU + (long long)(kk + 64) * kF, lds_off(&Bul[nxt][0]), 64, kF);
    }
    const signed char* Ab = Alds[cur];
    const signed char* Gb = Bgl[cur];
    const signed char* Ub = Bul[cur];
#else
    __syncthreads();
    load_a_tile128(gA + kk, kD, Alds[0], tid);
    load_b_tile64(gG + (long long)kk * kF, kF, Bgl[0], tid);
    load_b_tile64(gU + (long long)kk * kF, kF, Bul[0], tid);
    __syncthreads();
    const signed char* Ab = Alds[0];
    const signed char* Gb = Bgl[0];
    const signed char* Ub = Bul[0];
#endif
    v8i a = frag_a(Ab, lane, wave * 16);
#pragma unroll
    for (int t = 0; t < 4; t++) {
      v8i bg = frag_b(Gb, lane, t * 16);
      v8i bu = frag_b(Ub, lane, t * 16);
      accg[t] = __builtin_amdgcn_wmma_i32_16x16x64_iu8(true, a, true, bg,
                                                       accg[t], false, false);
      accu[t] = __builtin_amdgcn_wmma_i32_16x16x64_iu8(true, a, true, bu,
                                                       accu[t], false, false);
    }
#if USE_TDM
    if (more && wave < 3) __builtin_amdgcn_s_wait_tensorcnt(0);
    __syncthreads();
    cur = nxt;
#endif
  }

  const float wg = wgs[e], wu = wus[e];
  const int mbase = wave * 16 + (lane >> 4) * 8;
  const int ncol = lane & 15;
  float rs[8];
#pragma unroll
  for (int r = 0; r < 8; r++)
    rs[r] = 1.f / xs[(long long)e * kC + m0 + mbase + r];
#pragma unroll
  for (int t = 0; t < 4; t++) {
#pragma unroll
    for (int r = 0; r < 8; r++) {
      const int m = mbase + r;
      const int n = t * 16 + ncol;
      float g = (float)accg[t][r] * wg * rs[r];
      float u = (float)accu[t][r] * wu * rs[r];
      float hv = u / (1.f + __expf(-g));
      h[((long long)e * kC + m0 + m) * kF + n0 + n] = hv;
    }
  }
}

__global__ void __launch_bounds__(256) ffn_down_kernel(
    const signed char* __restrict__ hq, const float* __restrict__ hs,
    const signed char* __restrict__ wdq, const float* __restrict__ wds,
    float* __restrict__ out) {
#if USE_TDM
  __shared__ signed char Alds[2][128 * 64];
  __shared__ signed char Bdl[2][64 * 64];
#else
  __shared__ signed char Alds[1][128 * 64];
  __shared__ signed char Bdl[1][64 * 64];
#endif

  const int e = blockIdx.z;
  const int m0 = blockIdx.x * 128;
  const int n0 = blockIdx.y * 64;
  const int tid = threadIdx.x;
  const int lane = tid & 31;
  const int wave = tid >> 5;

  const signed char* gA = hq + ((long long)e * kC + m0) * kF;
  const signed char* gB = wdq + (long long)e * kF * kD + n0;

  v8i acc[4];
#pragma unroll
  for (int t = 0; t < 4; t++) {
    v8i z = {};
    acc[t] = z;
  }

#if USE_TDM
  if (wave == 0) tdm_load_2d(gA, lds_off(&Alds[0][0]), 128, kF);
  else if (wave == 1) tdm_load_2d(gB, lds_off(&Bdl[0][0]), 64, kD);
  if (wave < 2) __builtin_amdgcn_s_wait_tensorcnt(0);
  __syncthreads();
  int cur = 0;
#endif

  for (int kk = 0; kk < kF; kk += 64) {
#if USE_TDM
    const int nxt = cur ^ 1;
    const bool more = (kk + 64) < kF;
    if (more) {
      if (wave == 0)
        tdm_load_2d(gA + kk + 64, lds_off(&Alds[nxt][0]), 128, kF);
      else if (wave == 1)
        tdm_load_2d(gB + (long long)(kk + 64) * kD, lds_off(&Bdl[nxt][0]), 64, kD);
    }
    const signed char* Ab = Alds[cur];
    const signed char* Bb = Bdl[cur];
#else
    __syncthreads();
    load_a_tile128(gA + kk, kF, Alds[0], tid);
    load_b_tile64(gB + (long long)kk * kD, kD, Bdl[0], tid);
    __syncthreads();
    const signed char* Ab = Alds[0];
    const signed char* Bb = Bdl[0];
#endif
    v8i a = frag_a(Ab, lane, wave * 16);
#pragma unroll
    for (int t = 0; t < 4; t++) {
      v8i b = frag_b(Bb, lane, t * 16);
      acc[t] = __builtin_amdgcn_wmma_i32_16x16x64_iu8(true, a, true, b, acc[t],
                                                      false, false);
    }
#if USE_TDM
    if (more && wave < 2) __builtin_amdgcn_s_wait_tensorcnt(0);
    __syncthreads();
    cur = nxt;
#endif
  }

  const float wd = wds[e];
  const int mbase = wave * 16 + (lane >> 4) * 8;
  const int ncol = lane & 15;
  float rs[8];
#pragma unroll
  for (int r = 0; r < 8; r++)
    rs[r] = 1.f / hs[(long long)e * kC + m0 + mbase + r];
#pragma unroll
  for (int t = 0; t < 4; t++) {
#pragma unroll
    for (int r = 0; r < 8; r++) {
      const int m = mbase + r;
      const int n = t * 16 + ncol;
      out[((long long)e * kC + m0 + m) * kD + n0 + n] =
          (float)acc[t][r] * wd * rs[r];
    }
  }
}

extern "C" void kernel_launch(void* const* d_in, const int* in_sizes, int n_in,
                              void* d_out, int out_size, void* d_ws,
                              size_t ws_size, hipStream_t stream) {
  const float* x = (const float*)d_in[0];
  const float* w_gate = (const float*)d_in[1];
  const float* w_up = (const float*)d_in[2];
  const float* w_down = (const float*)d_in[3];
  float* out = (float*)d_out;

  const long long nWE = (long long)kF * kD;

  char* p = (char*)d_ws;
  signed char* xq = (signed char*)p;  p += (size_t)kE * kC * kD;
  signed char* wgq = (signed char*)p; p += (size_t)kE * nWE;
  signed char* wuq = (signed char*)p; p += (size_t)kE * nWE;
  signed char* wdq = (signed char*)p; p += (size_t)kE * nWE;
  signed char* hq = (signed char*)p;  p += (size_t)kE * kC * kF;
  float* h = (float*)p;   p += (size_t)kE * kC * kF * 4;
  float* xs = (float*)p;  p += (size_t)kE * kC * 4;
  float* hs = (float*)p;  p += (size_t)kE * kC * 4;
  float* wgs = (float*)p; p += 256;
  float* wus = (float*)p; p += 256;
  float* wds = (float*)p; p += 256;

  wscale_kernel<<<kE, 256, 0, stream>>>(w_gate, wgs, nWE);
  wscale_kernel<<<kE, 256, 0, stream>>>(w_up, wus, nWE);
  wscale_kernel<<<kE, 256, 0, stream>>>(w_down, wds, nWE);

  dim3 tg_gu(kD / 64, kF / 64, kE);
  wquant_t_kernel<<<tg_gu, 256, 0, stream>>>(w_gate, wgs, wgq, kF, kD);
  wquant_t_kernel<<<tg_gu, 256, 0, stream>>>(w_up, wus, wuq, kF, kD);
  dim3 tg_d(kF / 64, kD / 64, kE);
  wquant_t_kernel<<<tg_d, 256, 0, stream>>>(w_down, wds, wdq, kD, kF);

  aquant_kernel<<<kE * kC, 256, 0, stream>>>(x, xq, xs, kD);

  dim3 g1(kC / 128, kF / 64, kE);
  ffn_gateup_kernel<<<g1, 256, 0, stream>>>(xq, xs, wgq, wuq, wgs, wus, h);

  aquant_kernel<<<kE * kC, 256, 0, stream>>>(h, hq, hs, kF);

  dim3 g2(kC / 128, kD / 64, kE);
  ffn_down_kernel<<<g2, 256, 0, stream>>>(hq, hs, wdq, wds, out);
}
